// CausalMultiHeadsAttention_58042188038651
// MI455X (gfx1250) — compile-verified, matrix operand order checked by reading, not run on hardware
//
#include <hip/hip_runtime.h>
#include <math.h>

// Problem constants (match reference)
#define D_MODEL   1024
#define NUM_HEADS 16
#define HEAD_DIM  64
#define SEQ       2048
#define BATCH     2
#define M_TOTAL   (BATCH * SEQ)   // 4096 rows

typedef __attribute__((ext_vector_type(16))) __bf16 v16bf;
typedef __attribute__((ext_vector_type(8)))  float  v8f;

__device__ __forceinline__ unsigned short f2bf(float f) {
  unsigned int u = __float_as_uint(f);
  u += 0x7FFFu + ((u >> 16) & 1u);          // round-to-nearest-even
  return (unsigned short)(u >> 16);
}

union FragU { v16bf bf; uint4 q[2]; unsigned short u[16]; };

// Load a 16x32 bf16 A-fragment (or 32x16 B-fragment) whose contraction dim is
// contiguous in memory. rowk points at (row, k0). Lane layout per ISA 7.12.2:
// lanes 0-15: K = kb+0..7 (elems 0..7), kb+16..23 (elems 8..15), kb = (lane>=16)*8.
__device__ __forceinline__ v16bf frag_ld(const unsigned short* rowk, int hi) {
  const int kb = hi << 3;
  FragU u;
  u.q[0] = *(const uint4*)(rowk + kb);
  u.q[1] = *(const uint4*)(rowk + kb + 16);
  return u.bf;
}

__device__ __forceinline__ v8f wmma_bf16(v16bf a, v16bf b, v8f c) {
  // D = A(16x32 bf16) * B(32x16 bf16) + C(16x16 f32)
  return __builtin_amdgcn_wmma_f32_16x16x32_bf16(false, a, false, b, (short)0, c,
                                                 false, false);
}

// Low 32 bits of a flat shared-memory address are the LDS byte offset
// (ISA 10.2).
__device__ __forceinline__ unsigned lds_off(const void* p) {
  return (unsigned)(size_t)p;
}

// gfx1250 async copy global -> LDS, 16 bytes per lane, tracked by ASYNCcnt.
__device__ __forceinline__ void async_b128(unsigned ldsoff,
                                           const unsigned short* gptr) {
  asm volatile("global_load_async_to_lds_b128 %0, %1, off"
               :
               : "v"(ldsoff), "v"((unsigned long long)(size_t)gptr)
               : "memory");
}

// ---------------------------------------------------------------------------
// Kernel 0: fp32 -> bf16 conversion
// ---------------------------------------------------------------------------
__global__ void cvt_bf16_kernel(const float* __restrict__ src,
                                unsigned short* __restrict__ dst, int n) {
  int i = blockIdx.x * blockDim.x + threadIdx.x;
  if (i < n) dst[i] = f2bf(src[i]);
}

// ---------------------------------------------------------------------------
// Kernel 1: fused QKV projection + RoPE (+ V transposed store)
// grid: (M_TOTAL/32, NUM_HEADS, 3), block: 32 (one wave -> 32x64 output tile)
// q[b,h,l,e] = sum_d x[b,l,d] * W[h*64+e, d]
// ---------------------------------------------------------------------------
__global__ __launch_bounds__(32) void qkv_rope_kernel(
    const unsigned short* __restrict__ xb,   // [M_TOTAL, D_MODEL] bf16
    const unsigned short* __restrict__ wq,   // [D_MODEL, D_MODEL] bf16
    const unsigned short* __restrict__ wk,
    const unsigned short* __restrict__ wv,
    const int* __restrict__ tp,              // [SEQ]
    unsigned short* __restrict__ qb,         // [B,H,SEQ,64] bf16
    unsigned short* __restrict__ kb,         // [B,H,SEQ,64] bf16
    unsigned short* __restrict__ vT)         // [B,H,64,SEQ] bf16
{
  const int lane = threadIdx.x & 31;
  const int hi = lane >> 4, ln = lane & 15;
  const int mt = blockIdx.x;      // row tile (32 rows)
  const int h  = blockIdx.y;      // head == 64-col group
  const int p  = blockIdx.z;      // 0=Q 1=K 2=V
  const unsigned short* W = (p == 0) ? wq : (p == 1) ? wk : wv;
  const int mbase = mt * 32;

  const unsigned short* arow0 = xb + (size_t)(mbase + ln) * D_MODEL;
  const unsigned short* arow1 = xb + (size_t)(mbase + 16 + ln) * D_MODEL;

  v8f acc[2][4];
#pragma unroll
  for (int rt = 0; rt < 2; ++rt)
#pragma unroll
    for (int t = 0; t < 4; ++t)
#pragma unroll
      for (int r = 0; r < 8; ++r) acc[rt][t][r] = 0.0f;

  for (int k0 = 0; k0 < D_MODEL; k0 += 32) {
    v16bf a0 = frag_ld(arow0 + k0, hi);
    v16bf a1 = frag_ld(arow1 + k0, hi);
#pragma unroll
    for (int t = 0; t < 4; ++t) {
      const unsigned short* wrow =
          W + (size_t)(h * 64 + t * 16 + ln) * D_MODEL + k0;
      v16bf w = frag_ld(wrow, hi);
      acc[0][t] = wmma_bf16(a0, w, acc[0][t]);
      acc[1][t] = wmma_bf16(a1, w, acc[1][t]);
    }
  }

  // Spill tile to LDS so RoPE can pair adjacent columns.
  __shared__ float sacc[32 * 64];
#pragma unroll
  for (int rt = 0; rt < 2; ++rt)
#pragma unroll
    for (int t = 0; t < 4; ++t)
#pragma unroll
      for (int r = 0; r < 8; ++r)
        sacc[(rt * 16 + r + hi * 8) * 64 + t * 16 + ln] = acc[rt][t][r];
  __syncthreads();

  if (p < 2) {
    unsigned short* dst = (p == 0) ? qb : kb;
    for (int idx = lane; idx < 1024; idx += 32) {  // 32 rows * 32 pairs
      const int row = idx >> 5, j = idx & 31;
      const float x0 = sacc[row * 64 + 2 * j];
      const float x1 = sacc[row * 64 + 2 * j + 1];
      const int m = mbase + row;
      const int b = m >> 11;             // m / SEQ
      const int l = m & (SEQ - 1);
      const float pos = (float)tp[l];
      const float ang = pos * __powf(10000.0f, -(float)j * (1.0f / 32.0f));
      float s, c;
      __sincosf(ang, &s, &c);
      unsigned short* o =
          dst + ((size_t)(b * NUM_HEADS + h) * SEQ + l) * HEAD_DIM;
      o[2 * j]     = f2bf(x0 * c - x1 * s);
      o[2 * j + 1] = f2bf(x1 * c + x0 * s);
    }
  } else {
    for (int idx = lane; idx < 2048; idx += 32) {  // 32 rows * 64 cols
      const int row = idx >> 6, e = idx & 63;
      const int m = mbase + row;
      const int b = m >> 11;
      const int l = m & (SEQ - 1);
      vT[((size_t)(b * NUM_HEADS + h) * HEAD_DIM + e) * SEQ + l] =
          f2bf(sacc[row * 64 + e]);
    }
  }
}

// ---------------------------------------------------------------------------
// Kernel 2: causal flash attention, one wave per (b,h, 16-row query tile).
// K/V tiles are double-buffered in LDS via async global->LDS copies
// (ASYNCcnt), prefetched one 32-key step ahead.
// grid: (SEQ/16, BATCH*NUM_HEADS), block: 32
// ---------------------------------------------------------------------------
__global__ __launch_bounds__(32) void attn_kernel(
    const unsigned short* __restrict__ qbuf,  // [B,H,SEQ,64]
    const unsigned short* __restrict__ kbuf,  // [B,H,SEQ,64]
    const unsigned short* __restrict__ vT,    // [B,H,64,SEQ]
    unsigned short* __restrict__ ab)          // [B,SEQ,D_MODEL] bf16
{
  const int lane = threadIdx.x & 31;
  const int hi = lane >> 4, ln = lane & 15;
  const int it = blockIdx.x;
  const int bh = blockIdx.y;
  const int b = bh >> 4, h = bh & 15;
  const int qb0 = it * 16;

  __shared__ unsigned short kT[2][32 * 64];   // 32 keys x 64 e, row-major
  __shared__ unsigned short vS[2][64 * 32];   // 64 e x 32 keys, row-major
  __shared__ unsigned short pLDS[16 * 32];

  const unsigned short* kbase = kbuf + (size_t)bh * SEQ * HEAD_DIM;
  const unsigned short* vbase = vT + (size_t)bh * HEAD_DIM * SEQ;

  const unsigned koff[2] = {lds_off(&kT[0][0]), lds_off(&kT[1][0])};
  const unsigned voff[2] = {lds_off(&vS[0][0]), lds_off(&vS[1][0])};

  // --- async staging of one 32-key tile: 8 + 8 = 16 async instructions ---
  auto stage = [&](int jb, int buf) {
    const unsigned short* gk = kbase + (size_t)jb * HEAD_DIM;  // 4KB contiguous
#pragma unroll
    for (int c = 0; c < 8; ++c) {
      const unsigned idx = (unsigned)(c * 32 + lane);
      async_b128(koff[buf] + idx * 16u, gk + idx * 8);
    }
#pragma unroll
    for (int half = 0; half < 2; ++half) {
      const int e = lane + half * 32;
      const unsigned short* gv = vbase + (size_t)e * SEQ + jb;
#pragma unroll
      for (int qc = 0; qc < 4; ++qc)
        async_b128(voff[buf] + (unsigned)(e * 32 + qc * 8) * 2u, gv + qc * 8);
    }
  };

  const unsigned short* qrow =
      qbuf + ((size_t)bh * SEQ + qb0 + ln) * HEAD_DIM;
  const v16bf qa0 = frag_ld(qrow, hi);
  const v16bf qa1 = frag_ld(qrow + 32, hi);

  v8f o[4];
  float mrow[8], lrow[8];
#pragma unroll
  for (int r = 0; r < 8; ++r) { mrow[r] = -INFINITY; lrow[r] = 0.0f; }
#pragma unroll
  for (int t = 0; t < 4; ++t)
#pragma unroll
    for (int r = 0; r < 8; ++r) o[t][r] = 0.0f;

  const int jmax = (qb0 + 15) >> 5;
  stage(0, 0);                                   // prologue: tile 0 in flight

  for (int j = 0; j <= jmax; ++j) {
    const int jb = j * 32;
    const int buf = j & 1;

    if (j < jmax) {
      stage(jb + 32, buf ^ 1);                   // prefetch next tile
      // 32 outstanding -> wait until only the 16 prefetch ops remain.
      asm volatile("s_wait_asynccnt 0x10" ::: "memory");
    } else {
      asm volatile("s_wait_asynccnt 0x0" ::: "memory");
    }

    v8f s0, s1;
#pragma unroll
    for (int r = 0; r < 8; ++r) { s0[r] = 0.0f; s1[r] = 0.0f; }

    // S = q . k^T from the LDS-staged K tile (K-dim = 64 -> 2 steps per tile)
    const unsigned short* krow0 = &kT[buf][ln * 64];
    const unsigned short* krow1 = &kT[buf][(16 + ln) * 64];
    s0 = wmma_bf16(qa0, frag_ld(krow0, hi), s0);
    s0 = wmma_bf16(qa1, frag_ld(krow0 + 32, hi), s0);
    s1 = wmma_bf16(qa0, frag_ld(krow1, hi), s1);
    s1 = wmma_bf16(qa1, frag_ld(krow1 + 32, hi), s1);

    const bool needmask = (jb + 31) > qb0;
#pragma unroll
    for (int r = 0; r < 8; ++r) {
      const int rowg = qb0 + r + hi * 8;
      float a0 = s0[r] * 0.125f;   // 1/sqrt(64)
      float a1 = s1[r] * 0.125f;
      if (needmask) {
        if (jb + ln > rowg)      a0 = -INFINITY;
        if (jb + 16 + ln > rowg) a1 = -INFINITY;
      }
      float mx = fmaxf(a0, a1);
#pragma unroll
      for (int d = 1; d < 16; d <<= 1) mx = fmaxf(mx, __shfl_xor(mx, d, 16));
      const float mn = fmaxf(mrow[r], mx);
      const float p0 = __expf(a0 - mn);
      const float p1 = __expf(a1 - mn);
      float sm = p0 + p1;
#pragma unroll
      for (int d = 1; d < 16; d <<= 1) sm += __shfl_xor(sm, d, 16);
      const float alpha = __expf(mrow[r] - mn);
      lrow[r] = lrow[r] * alpha + sm;
      mrow[r] = mn;
#pragma unroll
      for (int t = 0; t < 4; ++t) o[t][r] *= alpha;
      s0[r] = p0;
      s1[r] = p1;
    }

    // Re-layout P (D-fragment f32) -> A-fragment bf16 via LDS.
#pragma unroll
    for (int r = 0; r < 8; ++r) {
      pLDS[(r + hi * 8) * 32 + ln]      = f2bf(s0[r]);
      pLDS[(r + hi * 8) * 32 + 16 + ln] = f2bf(s1[r]);
    }
    __syncthreads();
    FragU up;
    {
      const unsigned short* prow = pLDS + ln * 32 + hi * 8;
#pragma unroll
      for (int i = 0; i < 8; ++i) {
        up.u[i]     = prow[i];
        up.u[8 + i] = prow[16 + i];
      }
    }
    // O += P . V from the LDS-staged V^T tile (rows contiguous along keys)
#pragma unroll
    for (int t = 0; t < 4; ++t) {
      const unsigned short* vrow = &vS[buf][(t * 16 + ln) * 32];
      o[t] = wmma_bf16(up.bf, frag_ld(vrow, hi), o[t]);
    }
    __syncthreads();   // protect pLDS before next iteration
  }

  // Epilogue: normalize and store to [b, l, h*64 + e] bf16.
#pragma unroll
  for (int r = 0; r < 8; ++r) {
    const float inv = 1.0f / lrow[r];
    const int rowg = qb0 + r + hi * 8;
    unsigned short* orow =
        ab + ((size_t)(b * SEQ + rowg)) * D_MODEL + h * HEAD_DIM;
#pragma unroll
    for (int t = 0; t < 4; ++t) orow[t * 16 + ln] = f2bf(o[t][r] * inv);
  }
}

// ---------------------------------------------------------------------------
// Kernel 3: output projection  out = attn @ O_w^T  (f32 result)
// grid: (M_TOTAL/32, D_MODEL/64), block: 32 (one wave -> 32x64 tile)
// ---------------------------------------------------------------------------
__global__ __launch_bounds__(32) void oproj_kernel(
    const unsigned short* __restrict__ ab,  // [M_TOTAL, D_MODEL] bf16
    const unsigned short* __restrict__ wo,  // [D_MODEL, D_MODEL] bf16
    float* __restrict__ out)                // [M_TOTAL, D_MODEL] f32
{
  const int lane = threadIdx.x & 31;
  const int hi = lane >> 4, ln = lane & 15;
  const int mt = blockIdx.x;
  const int g  = blockIdx.y;
  const int mbase = mt * 32;

  const unsigned short* arow0 = ab + (size_t)(mbase + ln) * D_MODEL;
  const unsigned short* arow1 = ab + (size_t)(mbase + 16 + ln) * D_MODEL;

  v8f acc[2][4];
#pragma unroll
  for (int rt = 0; rt < 2; ++rt)
#pragma unroll
    for (int t = 0; t < 4; ++t)
#pragma unroll
      for (int r = 0; r < 8; ++r) acc[rt][t][r] = 0.0f;

  for (int k0 = 0; k0 < D_MODEL; k0 += 32) {
    v16bf a0 = frag_ld(arow0 + k0, hi);
    v16bf a1 = frag_ld(arow1 + k0, hi);
#pragma unroll
    for (int t = 0; t < 4; ++t) {
      const unsigned short* wrow =
          wo + (size_t)(g * 64 + t * 16 + ln) * D_MODEL + k0;
      v16bf w = frag_ld(wrow, hi);
      acc[0][t] = wmma_bf16(a0, w, acc[0][t]);
      acc[1][t] = wmma_bf16(a1, w, acc[1][t]);
    }
  }

#pragma unroll
  for (int rt = 0; rt < 2; ++rt)
#pragma unroll
    for (int t = 0; t < 4; ++t)
#pragma unroll
      for (int r = 0; r < 8; ++r)
        out[(size_t)(mbase + rt * 16 + r + hi * 8) * D_MODEL + g * 64 +
            t * 16 + ln] = acc[rt][t][r];
}

// ---------------------------------------------------------------------------
extern "C" void kernel_launch(void* const* d_in, const int* in_sizes, int n_in,
                              void* d_out, int out_size, void* d_ws,
                              size_t ws_size, hipStream_t stream) {
  const float* x  = (const float*)d_in[0];
  const int*   tp = (const int*)d_in[1];
  const float* Q  = (const float*)d_in[2];
  const float* K  = (const float*)d_in[3];
  const float* V  = (const float*)d_in[4];
  const float* Ow = (const float*)d_in[5];
  float* out = (float*)d_out;

  const size_t XE = (size_t)M_TOTAL * D_MODEL;                   // 4,194,304
  const size_t WE = (size_t)D_MODEL * D_MODEL;                   // 1,048,576
  const size_t QE = (size_t)BATCH * NUM_HEADS * SEQ * HEAD_DIM;  // 4,194,304

  unsigned short* ws = (unsigned short*)d_ws;
  unsigned short* xb = ws;
  unsigned short* wq = xb + XE;
  unsigned short* wk = wq + WE;
  unsigned short* wv = wk + WE;
  unsigned short* wo = wv + WE;
  unsigned short* qb = wo + WE;
  unsigned short* kb = qb + QE;
  unsigned short* vT = kb + QE;
  unsigned short* ab = vT + QE;   // total 48 MiB of d_ws

  cvt_bf16_kernel<<<(int)((XE + 255) / 256), 256, 0, stream>>>(x, xb, (int)XE);
  cvt_bf16_kernel<<<(int)((WE + 255) / 256), 256, 0, stream>>>(Q, wq, (int)WE);
  cvt_bf16_kernel<<<(int)((WE + 255) / 256), 256, 0, stream>>>(K, wk, (int)WE);
  cvt_bf16_kernel<<<(int)((WE + 255) / 256), 256, 0, stream>>>(V, wv, (int)WE);
  cvt_bf16_kernel<<<(int)((WE + 255) / 256), 256, 0, stream>>>(Ow, wo, (int)WE);

  qkv_rope_kernel<<<dim3(M_TOTAL / 32, NUM_HEADS, 3), 32, 0, stream>>>(
      xb, wq, wk, wv, tp, qb, kb, vT);

  attn_kernel<<<dim3(SEQ / 16, BATCH * NUM_HEADS), 32, 0, stream>>>(
      qb, kb, vT, ab);

  oproj_kernel<<<dim3(M_TOTAL / 32, D_MODEL / 64), 32, 0, stream>>>(
      ab, wo, out);
}
